// SLIM_2018634629355
// MI455X (gfx1250) — matrix operand order repaired and checked by reading, not compiled, not run on hardware
//
#include <hip/hip_runtime.h>

// ---------------------------------------------------------------------------
// MI455X (gfx1250). Channel-mixing GEMMs on v_wmma_f32_16x16x32_bf16 (wave32),
// fp32 accumulate; fp32->bf16 conversion fused into LDS staging; epilogues
// (bias/GELU/gate-mul/residual/alpha) fused into the GEMM store.
// Block tile 64(o) x 64(n), K-step 64 -> 4 WMMA per wave per barrier pair.
// Constants: B=8, C=192, H=W=128, N=16384, GS={24,24,48,96}, RS={8,8,12,24}.
// ---------------------------------------------------------------------------

typedef __attribute__((ext_vector_type(16))) __bf16 v16bf;
typedef __attribute__((ext_vector_type(8)))  float  v8f;

#define NTOK 16384
#define CCH  192
#define BB   8

__device__ __forceinline__ unsigned short f2bf(float f) {
  unsigned u = __builtin_bit_cast(unsigned, f);
  u += 0x7FFFu + ((u >> 16) & 1u);          // round-to-nearest-even
  return (unsigned short)(u >> 16);
}

__device__ __forceinline__ unsigned pack_bf2(float lo, float hi) {
  return (unsigned)f2bf(lo) | ((unsigned)f2bf(hi) << 16);
}

__device__ __forceinline__ float gelu_f(float x) {
  return 0.5f * x * (1.0f + erff(x * 0.70710678118654752440f));
}

// ---------------------------------------------------------------------------
// WMMA GEMM: Y[b][o,n] = epilogue( sum_c W[o,c] * X[b][c,n] )
// 256 threads = 8 waves; tile 64(o) x 64(n); K-step 64.
// Wave w: osub=(w&3)*16, nsub=(w>>2)*32; two accumulators (n, n+16).
// GUARD=false requires O%64==0 && K%64==0 (no predication anywhere).
// modes: 0 plain | 1 +bias | 2 gelu(+bias) | 3 gelu(+bias)*aux |
//        4 +bias+aux | 5 aux + alpha*val
// ---------------------------------------------------------------------------
template <bool GUARD>
__global__ __launch_bounds__(256)
void gemm_wmma(const float* __restrict__ W, const float* __restrict__ X,
               float* __restrict__ Y, const float* __restrict__ bias,
               const float* __restrict__ aux, const float* __restrict__ alpha_p,
               int O, int K, int N, int mode)
{
  __shared__ unsigned short sA[64][66];   // [o][c]
  __shared__ unsigned short sB[64][66];   // [n][c] (transposed stage)

  const int tid  = threadIdx.x;
  const int lane = tid & 31, wv = tid >> 5;
  const int half = lane >> 4, l = lane & 15;
  const int osub = (wv & 3) * 16;
  const int nsub = (wv >> 2) * 32;
  const int n0 = blockIdx.x * 64;
  const int o0 = blockIdx.y * 64;
  const int b  = blockIdx.z;
  const float* __restrict__ Xb = X + (size_t)b * K * N;
  const size_t ybase = (size_t)b * O * N;

  v8f acc0 = {0.f,0.f,0.f,0.f,0.f,0.f,0.f,0.f};
  v8f acc1 = {0.f,0.f,0.f,0.f,0.f,0.f,0.f,0.f};

  // staging coordinates
  const int acd = tid & 31;           // A dword col (c = 2*acd)
  const int arw = tid >> 5;           // A row base, 8 rows per pass
  const int bn  = tid & 63;           // B n column
  const int bcp = tid >> 6;           // B c-pair base, 4 per pass

  const int ksteps = (K + 63) >> 6;
  for (int kt = 0; kt < ksteps; ++kt) {
    const int c0 = kt * 64;

    // ---- A tile 64x64: batch all loads first (unconditional, clamped) ----
    float a0[8], a1[8];
#pragma unroll
    for (int i = 0; i < 8; ++i) {
      const int row = o0 + arw + 8 * i;
      const int c   = c0 + 2 * acd;
      if (GUARD) {
        const int ro = row < O ? row : O - 1;
        const int cc = (c <= K - 2) ? c : (K - 2);
        const float* p = &W[(size_t)ro * K + cc];
        const float t0 = p[0], t1 = p[1];
        a0[i] = (row < O && c     < K) ? t0 : 0.f;
        a1[i] = (row < O && c + 1 < K) ? t1 : 0.f;
      } else {
        const float* p = &W[(size_t)row * K + c];
        a0[i] = p[0]; a1[i] = p[1];
      }
    }
#pragma unroll
    for (int i = 0; i < 8; ++i)
      *(unsigned*)&sA[arw + 8 * i][2 * acd] = pack_bf2(a0[i], a1[i]);

    // ---- B tile 64(c)x64(n), transposed into sB[n][c] ----
    float b0[8], b1[8];
#pragma unroll
    for (int i = 0; i < 8; ++i) {
      const int c = c0 + 2 * (bcp + 4 * i);
      const int n = n0 + bn;
      if (GUARD) {
        const int cA = (c     < K) ? c     : K - 1;
        const int cB = (c + 1 < K) ? c + 1 : K - 1;
        const float t0 = Xb[(size_t)cA * N + n];
        const float t1 = Xb[(size_t)cB * N + n];
        b0[i] = (c     < K) ? t0 : 0.f;
        b1[i] = (c + 1 < K) ? t1 : 0.f;
      } else {
        b0[i] = Xb[(size_t)c * N + n];
        b1[i] = Xb[(size_t)(c + 1) * N + n];
      }
    }
#pragma unroll
    for (int i = 0; i < 8; ++i)
      *(unsigned*)&sB[bn][2 * (bcp + 4 * i)] = pack_bf2(b0[i], b1[i]);

    // prefetch next K-slab of X (emits global_prefetch on gfx1250)
    if (kt + 1 < ksteps && (c0 + 64 + 2 * bcp) < K)
      __builtin_prefetch(&Xb[(size_t)(c0 + 64 + 2 * bcp) * N + n0 + bn], 0, 0);

    __syncthreads();

    // ---- 4 WMMAs per wave ----
#pragma unroll
    for (int ks = 0; ks < 64; ks += 32) {
      union { v16bf v; unsigned u[8]; } fa, fb0, fb1;
      const int arow = osub + l;
      const int br0 = nsub + l, br1 = nsub + 16 + l;
#pragma unroll
      for (int j = 0; j < 8; ++j) {
        // A 16x32 bf16 layout (ISA 7.12.2): lanes0-15 K={0..7,16..23}, +8 for hi
        const int kA = ks + ((j < 4) ? 2 * j : 16 + 2 * (j - 4)) + half * 8;
        fa.u[j]  = *(const unsigned*)&sA[arow][kA];
        // B 32x16 bf16 layout: same K order as A (lanes0-15 K={0..7,16..23}, +8 for hi)
        const int kB = ks + ((j < 4) ? 2 * j : 16 + 2 * (j - 4)) + half * 8;
        fb0.u[j] = *(const unsigned*)&sB[br0][kB];
        fb1.u[j] = *(const unsigned*)&sB[br1][kB];
      }
      acc0 = __builtin_amdgcn_wmma_f32_16x16x32_bf16(false, fa.v, false, fb0.v,
                                                     (short)0, acc0, false, false);
      acc1 = __builtin_amdgcn_wmma_f32_16x16x32_bf16(false, fa.v, false, fb1.v,
                                                     (short)0, acc1, false, false);
    }
    __syncthreads();
  }

  // ---- fused epilogue ----
  const float alpha = (mode == 5) ? alpha_p[0] : 0.f;
#pragma unroll
  for (int r = 0; r < 8; ++r) {
    const int m = r + half * 8;            // D layout: VGPR r -> M=r / r+8
    const int o = o0 + osub + m;
    if (!GUARD || o < O) {
#pragma unroll
      for (int t = 0; t < 2; ++t) {
        const int n = n0 + nsub + 16 * t + l;
        const size_t idx = ybase + (size_t)o * N + n;
        float v = (t == 0) ? acc0[r] : acc1[r];
        if      (mode == 1) v += bias[o];
        else if (mode == 2) v = gelu_f(v + bias[o]);
        else if (mode == 3) v = gelu_f(v + bias[o]) * aux[idx];
        else if (mode == 4) v = v + bias[o] + aux[idx];
        else if (mode == 5) v = aux[idx] + alpha * v;
        Y[idx] = v;
      }
    }
  }
}

// ---------------------------------------------------------------------------
// Depthwise 3x3 SAME conv. grid (H, CH, B), block W=128.
// ---------------------------------------------------------------------------
__global__ void dwconv3x3(const float* __restrict__ in, const float* __restrict__ wdw,
                          float* __restrict__ out, int CH, int chin_off, int CHIN_TOTAL)
{
  const int x = threadIdx.x;
  const int y = blockIdx.x;
  const int c = blockIdx.y;
  const int b = blockIdx.z;
  const float* ip = in + ((size_t)b * CHIN_TOTAL + chin_off + c) * NTOK;
  const float* wp = wdw + c * 9;
  float s = 0.f;
#pragma unroll
  for (int dy = -1; dy <= 1; ++dy) {
    int yy = y + dy;
    if (yy < 0 || yy >= 128) continue;
#pragma unroll
    for (int dx = -1; dx <= 1; ++dx) {
      int xx = x + dx;
      if (xx < 0 || xx >= 128) continue;
      s += wp[(dy + 1) * 3 + (dx + 1)] * ip[yy * 128 + xx];
    }
  }
  out[((size_t)b * CH + c) * NTOK + y * 128 + x] = s;
}

// mean over N. grid (C, B), block 256
__global__ void mean_n(const float* __restrict__ xv, float* __restrict__ ctx)
{
  __shared__ float red[256];
  int c = blockIdx.x, b = blockIdx.y, t = threadIdx.x;
  const float* p = xv + ((size_t)b * CCH + c) * NTOK;
  float s = 0.f;
  for (int n = t; n < NTOK; n += 256) s += p[n];
  red[t] = s; __syncthreads();
  for (int st = 128; st > 0; st >>= 1) { if (t < st) red[t] += red[t + st]; __syncthreads(); }
  if (t == 0) ctx[b * CCH + c] = red[0] * (1.f / (float)NTOK);
}

// rank projections Qr/Kr/Vr. grid (N/128, r, B), block 128
__global__ void rank_proj(const float* __restrict__ qkv, const float* __restrict__ wq,
                          const float* __restrict__ wk, const float* __restrict__ wv,
                          float* __restrict__ Qr, float* __restrict__ Kr, float* __restrict__ Vr,
                          int g, int c0, int r_off)
{
  int n = blockIdx.x * 128 + threadIdx.x;
  int r = blockIdx.y, b = blockIdx.z;
  const float* base = qkv + ((size_t)b * 576 + c0) * NTOK;
  float sq = 0.f, sk = 0.f, sv = 0.f;
  for (int c = 0; c < g; ++c) {
    float aq = wq[r * g + c], ak = wk[r * g + c], av = wv[r * g + c];
    sq += aq * base[(size_t)c * NTOK + n];
    sk += ak * base[(size_t)(192 + c) * NTOK + n];
    sv += av * base[(size_t)(384 + c) * NTOK + n];
  }
  size_t o = ((size_t)b * 52 + r_off + r) * NTOK + n;
  Qr[o] = sq; Kr[o] = sk; Vr[o] = sv;
}

// L2 norms over N. grid (52, B), block 256
__global__ void rank_norms(const float* __restrict__ Qr, const float* __restrict__ Kr,
                           float* __restrict__ norms)
{
  __shared__ float rq[256], rk[256];
  int r = blockIdx.x, b = blockIdx.y, t = threadIdx.x;
  const float* q = Qr + ((size_t)b * 52 + r) * NTOK;
  const float* k = Kr + ((size_t)b * 52 + r) * NTOK;
  float sq = 0.f, sk = 0.f;
  for (int n = t; n < NTOK; n += 256) { float a = q[n]; sq += a * a; float c = k[n]; sk += c * c; }
  rq[t] = sq; rk[t] = sk; __syncthreads();
  for (int st = 128; st > 0; st >>= 1) {
    if (t < st) { rq[t] += rq[t + st]; rk[t] += rk[t + st]; } __syncthreads();
  }
  if (t == 0) {
    norms[(b * 52 + r) * 2 + 0] = sqrtf(rq[0]);
    norms[(b * 52 + r) * 2 + 1] = sqrtf(rk[0]);
  }
}

// Gram matrix G[r,s] = <Qr_r, Kr_s>. grid (r*r, B), block 256
__global__ void gram_k(const float* __restrict__ Qr, const float* __restrict__ Kr,
                       float* __restrict__ gram, int r, int r_off, int gram_off)
{
  __shared__ float red[256];
  int rr = blockIdx.x, b = blockIdx.y, t = threadIdx.x;
  int ri = rr / r, si = rr % r;
  const float* q = Qr + ((size_t)b * 52 + r_off + ri) * NTOK;
  const float* k = Kr + ((size_t)b * 52 + r_off + si) * NTOK;
  float s = 0.f;
  for (int n = t; n < NTOK; n += 256) s += q[n] * k[n];
  red[t] = s; __syncthreads();
  for (int st = 128; st > 0; st >>= 1) { if (t < st) red[t] += red[t + st]; __syncthreads(); }
  if (t == 0) gram[b * 848 + gram_off + rr] = red[0];
}

// softmax attention + modulation MLP. grid (B), block 192
__global__ void att_mod(const float* __restrict__ gram, const float* __restrict__ norms,
                        const float* __restrict__ temp_all, const float* __restrict__ ctx,
                        const float* __restrict__ iw1, const float* __restrict__ ib1,
                        const float* __restrict__ iw2, const float* __restrict__ ib2,
                        float* __restrict__ att, float* __restrict__ scale,
                        float* __restrict__ shift,
                        int g, int c0, int r, int r_off, int gram_off, int hdim, int ti)
{
  __shared__ float h[32];
  int b = blockIdx.x, t = threadIdx.x;
  float temp = temp_all[ti];
  if (t < r) {
    float nq = fmaxf(norms[(b * 52 + r_off + t) * 2 + 0], 1e-12f);
    float row[24];
    float mx = -3.4e38f;
    for (int s = 0; s < r; ++s) {
      float nk = fmaxf(norms[(b * 52 + r_off + s) * 2 + 1], 1e-12f);
      float v = gram[b * 848 + gram_off + t * r + s] / (nq * nk) * temp;
      row[s] = v; mx = fmaxf(mx, v);
    }
    float den = 0.f;
    for (int s = 0; s < r; ++s) { row[s] = expf(row[s] - mx); den += row[s]; }
    float inv = 1.f / den;
    for (int s = 0; s < r; ++s) att[b * 848 + gram_off + t * r + s] = row[s] * inv;
  }
  if (t < hdim) {
    float s = ib1[t];
    for (int c = 0; c < g; ++c) s += iw1[t * g + c] * ctx[b * CCH + c0 + c];
    h[t] = gelu_f(s);
  }
  __syncthreads();
  if (t < 2 * g) {
    float s = ib2[t];
    for (int j = 0; j < hdim; ++j) s += iw2[t * hdim + j] * h[j];
    if (t < g) scale[b * CCH + c0 + t] = s;
    else       shift[b * CCH + c0 + (t - g)] = s;
  }
}

// out_r = att @ Vr. grid (N/128, r, B), block 128
__global__ void attv_k(const float* __restrict__ att, const float* __restrict__ Vr,
                       float* __restrict__ out_r, int r, int r_off, int gram_off)
{
  int n = blockIdx.x * 128 + threadIdx.x;
  int ri = blockIdx.y, b = blockIdx.z;
  const float* a = att + b * 848 + gram_off + ri * r;
  float s = 0.f;
  for (int si = 0; si < r; ++si) s += a[si] * Vr[((size_t)b * 52 + r_off + si) * NTOK + n];
  out_r[((size_t)b * 52 + r_off + ri) * NTOK + n] = s;
}

// out_g = wb @ out_r, modulation, fused = out + (q+k). grid (N/128, g, B)
__global__ void outfuse_k(const float* __restrict__ out_r, const float* __restrict__ wb,
                          const float* __restrict__ scale, const float* __restrict__ shift,
                          const float* __restrict__ qkv_dw,
                          float* __restrict__ out_all, float* __restrict__ fused,
                          int g, int c0, int r, int r_off)
{
  int n = blockIdx.x * 128 + threadIdx.x;
  int c = blockIdx.y, b = blockIdx.z;
  float s = 0.f;
  for (int ri = 0; ri < r; ++ri)
    s += wb[c * r + ri] * out_r[((size_t)b * 52 + r_off + ri) * NTOK + n];
  float sc = scale[b * CCH + c0 + c], sh = shift[b * CCH + c0 + c];
  float ov = s * (1.f + sc) + sh;
  size_t oi = ((size_t)b * CCH + c0 + c) * NTOK + n;
  out_all[oi] = ov;
  float qv = qkv_dw[((size_t)b * 576 + c0 + c) * NTOK + n];
  float kv = qkv_dw[((size_t)b * 576 + 192 + c0 + c) * NTOK + n];
  fused[oi] = ov + qv + kv;
}

// ---------------------------------------------------------------------------
extern "C" void kernel_launch(void* const* d_in, const int* in_sizes, int n_in,
                              void* d_out, int out_size, void* d_ws, size_t ws_size,
                              hipStream_t stream) {
  (void)in_sizes; (void)n_in; (void)out_size; (void)ws_size;
  const float* x        = (const float*)d_in[0];
  const float* w_qkv    = (const float*)d_in[1];
  const float* w_qkv_dw = (const float*)d_in[2];
  const float* w_proj   = (const float*)d_in[3];
  const float* dw_w     = (const float*)d_in[4];
  const float* pw_w     = (const float*)d_in[5];
  const float* pw_b     = (const float*)d_in[6];
  const float *wq[4], *wk[4], *wv[4], *wb[4], *iw1[4], *ib1[4], *iw2[4], *ib2[4];
  for (int i = 0; i < 4; ++i) {
    wq[i]  = (const float*)d_in[7 + i];
    wk[i]  = (const float*)d_in[11 + i];
    wv[i]  = (const float*)d_in[15 + i];
    wb[i]  = (const float*)d_in[19 + i];
    iw1[i] = (const float*)d_in[23 + i];
    ib1[i] = (const float*)d_in[27 + i];
    iw2[i] = (const float*)d_in[31 + i];
    ib2[i] = (const float*)d_in[35 + i];
  }
  const float* down_w = (const float*)d_in[39];
  const float* down_b = (const float*)d_in[40];
  const float* up_w   = (const float*)d_in[41];
  const float* up_b   = (const float*)d_in[42];
  const float* gate_w = (const float*)d_in[43];
  const float* gate_b = (const float*)d_in[44];
  const float* temp   = (const float*)d_in[45];
  const float* alpha  = (const float*)d_in[46];
  float* out = (float*)d_out;
  float* w = (float*)d_ws;

  // Workspace layout (floats). Region A [0, 75.5M) and B [75.5M, 151M) reused.
  const size_t PLANE = (size_t)CCH * NTOK;                 // 3,145,728
  float* qkv_lin = w;                                      // B*576*N
  float* qkv_dw  = w + (size_t)BB * 576 * NTOK;            // B*576*N
  float* xv_dw   = qkv_dw + (size_t)BB * 576 * NTOK;       // B*C*N
  float* xv      = xv_dw + (size_t)BB * PLANE;             // B*C*N
  float* ctx     = xv + (size_t)BB * PLANE;                // B*C
  float* Qr      = ctx + BB * CCH;                         // B*52*N
  float* Kr      = Qr + (size_t)BB * 52 * NTOK;
  float* Vr      = Kr + (size_t)BB * 52 * NTOK;
  float* norms   = Vr + (size_t)BB * 52 * NTOK;            // B*52*2
  float* gram    = norms + BB * 52 * 2;                    // B*848
  float* attb    = gram + BB * 848;                        // B*848
  float* scaleb  = attb + BB * 848;                        // B*C
  float* shiftb  = scaleb + BB * CCH;                      // B*C
  float* out_r   = shiftb + BB * CCH;                      // B*52*N
  // Region A reuse (qkv_lin dead after dwconv):
  float* out_all = w;                                      // B*C*N
  float* fusedb  = w + (size_t)BB * PLANE;                 // B*C*N
  float* gatedb  = w + (size_t)2 * BB * PLANE;             // B*C*N
  // Region B reuse (qkv_dw dead after outfuse):
  float* tdown   = w + (size_t)BB * 576 * NTOK;            // B*48*N
  float* out2    = tdown + (size_t)BB * 48 * NTOK;         // B*C*N

  static const int GSh[4]  = {24, 24, 48, 96};
  static const int RSh[4]  = {8, 8, 12, 24};
  static const int C0[4]   = {0, 24, 48, 96};
  static const int ROFF[4] = {0, 8, 16, 28};
  static const int GOFF[4] = {0, 64, 128, 272};
  static const int HD[4]   = {6, 6, 12, 24};

  // 1) qkv = w_qkv @ x   (WMMA, mode 0, O=576 K=192 -> clean path)
  gemm_wmma<false><<<dim3(NTOK / 64, 9, BB), 256, 0, stream>>>(
      w_qkv, x, qkv_lin, nullptr, nullptr, nullptr, 576, 192, NTOK, 0);
  // 2) depthwise 3x3 on qkv
  dwconv3x3<<<dim3(128, 576, BB), 128, 0, stream>>>(qkv_lin, w_qkv_dw, qkv_dw, 576, 0, 576);
  // 3) xv_dw = dw3x3(v, dw_w)
  dwconv3x3<<<dim3(128, 192, BB), 128, 0, stream>>>(qkv_dw, dw_w, xv_dw, 192, 384, 576);
  // 4) xv = gelu(pw_w @ xv_dw + pw_b)   (WMMA, mode 2)
  gemm_wmma<false><<<dim3(NTOK / 64, 3, BB), 256, 0, stream>>>(
      pw_w, xv_dw, xv, pw_b, nullptr, nullptr, 192, 192, NTOK, 2);
  // 5) ctx_full = mean_N(xv)
  mean_n<<<dim3(CCH, BB), 256, 0, stream>>>(xv, ctx);
  // 6) per-group low-rank attention
  for (int i = 0; i < 4; ++i)
    rank_proj<<<dim3(NTOK / 128, RSh[i], BB), 128, 0, stream>>>(
        qkv_dw, wq[i], wk[i], wv[i], Qr, Kr, Vr, GSh[i], C0[i], ROFF[i]);
  rank_norms<<<dim3(52, BB), 256, 0, stream>>>(Qr, Kr, norms);
  for (int i = 0; i < 4; ++i)
    gram_k<<<dim3(RSh[i] * RSh[i], BB), 256, 0, stream>>>(
        Qr, Kr, gram, RSh[i], ROFF[i], GOFF[i]);
  for (int i = 0; i < 4; ++i)
    att_mod<<<dim3(BB), 192, 0, stream>>>(
        gram, norms, temp, ctx, iw1[i], ib1[i], iw2[i], ib2[i],
        attb, scaleb, shiftb, GSh[i], C0[i], RSh[i], ROFF[i], GOFF[i], HD[i], i);
  for (int i = 0; i < 4; ++i)
    attv_k<<<dim3(NTOK / 128, RSh[i], BB), 128, 0, stream>>>(
        attb, Vr, out_r, RSh[i], ROFF[i], GOFF[i]);
  for (int i = 0; i < 4; ++i)
    outfuse_k<<<dim3(NTOK / 128, GSh[i], BB), 128, 0, stream>>>(
        out_r, wb[i], scaleb, shiftb, qkv_dw, out_all, fusedb,
        GSh[i], C0[i], RSh[i], ROFF[i]);
  // 7) gated = gelu(gate_w @ fused + gate_b) * fused   (WMMA, mode 3)
  gemm_wmma<false><<<dim3(NTOK / 64, 3, BB), 256, 0, stream>>>(
      gate_w, fusedb, gatedb, gate_b, fusedb, nullptr, 192, 192, NTOK, 3);
  // 8) tdown = down_w @ gated + down_b   (WMMA, mode 1, O=48 -> guarded)
  gemm_wmma<true><<<dim3(NTOK / 64, 1, BB), 256, 0, stream>>>(
      down_w, gatedb, tdown, down_b, nullptr, nullptr, 48, 192, NTOK, 1);
  // 9) out2 = out_all + (up_w @ tdown + up_b)   (WMMA, mode 4, K=48 -> guarded)
  gemm_wmma<true><<<dim3(NTOK / 64, 3, BB), 256, 0, stream>>>(
      up_w, tdown, out2, up_b, out_all, nullptr, 192, 48, NTOK, 4);
  // 10) out = x + alpha * (w_proj @ out2)   (WMMA, mode 5) -> d_out
  gemm_wmma<false><<<dim3(NTOK / 64, 3, BB), 256, 0, stream>>>(
      w_proj, out2, out, nullptr, x, alpha, 192, 192, NTOK, 5);
}
